// Attention_18786186952997
// MI455X (gfx1250) — compile-verified, matrix operand order checked by reading, not run on hardware
//
#include <hip/hip_runtime.h>
#include <hip/hip_bf16.h>

typedef __attribute__((ext_vector_type(16))) _Float16 v16h;
typedef __attribute__((ext_vector_type(8)))  _Float16 v8h;
typedef __attribute__((ext_vector_type(8)))  float    v8f;
typedef __attribute__((ext_vector_type(4)))  unsigned int v4u;
typedef __attribute__((ext_vector_type(8)))  int      v8i;
typedef __attribute__((ext_vector_type(4)))  int      v4i;

#if __has_builtin(__builtin_amdgcn_tensor_load_to_lds)
#define HAVE_TDM 1
#else
#define HAVE_TDM 0
#endif

// Build a 16-wide f16 A/B fragment from two contiguous 16-byte chunks.
// 16-bit WMMA operand layout (CDNA5 ISA 7.12.2): lanes 0-15 hold K=[0..7] in
// VGPR0-3 and K=[16..23] in VGPR4-7; lanes 16-31 hold K=[8..15] / K=[24..31].
__device__ __forceinline__ v16h mk16(const _Float16* lo, const _Float16* hi) {
    v8h a = *(const v8h*)lo;
    v8h b = *(const v8h*)hi;
    v16h r;
#pragma unroll
    for (int i = 0; i < 8; ++i) { r[i] = a[i]; r[i + 8] = b[i]; }
    return r;
}

__device__ __forceinline__ v8f wmma_f16(v16h a, v16h b, v8f c) {
    return __builtin_amdgcn_wmma_f32_16x16x32_f16(
        /*neg_a=*/false, a, /*neg_b=*/false, b,
        /*c_mod=*/(short)0, c, /*reuse_a=*/false, /*reuse_b=*/false);
}

#if HAVE_TDM
// Issue a TDM 2D tile load: `rows` x 32 f16 tile whose memory row stride is
// `d0_stride` elements, landing at LDS byte offset `lds_off`.  LDS pad feature:
// after every 16 dwords (one 64B row) insert 4 dwords -> 80B row stride, which
// keeps rows 16B-aligned for ds_load_b128 and is bank-conflict free.
__device__ __forceinline__ void tdm_load_tile(unsigned lds_off, const void* gptr,
                                              unsigned rows,
                                              unsigned long long d0_stride) {
    unsigned long long ga = (unsigned long long)(uintptr_t)gptr;
    v4u g0;
    g0[0] = 1u;                                            // count=1 (valid D#)
    g0[1] = lds_off;                                       // lds_addr (bytes)
    g0[2] = (unsigned)(ga & 0xffffffffu);                  // global_addr lo
    g0[3] = (unsigned)((ga >> 32) & 0x1ffffffu) | (2u << 30); // addr hi | type=2
    v8i g1;
    // data_size=1(2B)<<16 | pad_enable<<20 | pad_interval=3(16dw)<<22 | pad_amount=3(4dw)<<25
    g1[0] = (int)((1u << 16) | (1u << 20) | (3u << 22) | (3u << 25));
    g1[1] = (int)((32u & 0xffffu) << 16);                  // tensor_dim0[15:0]=32
    g1[2] = (int)(((32u >> 16) & 0xffffu) | ((rows & 0xffffu) << 16)); // dim0 hi | tensor_dim1 lo
    g1[3] = (int)(((rows >> 16) & 0xffffu) | (32u << 16)); // tensor_dim1 hi | tile_dim0=32
    g1[4] = (int)(rows & 0xffffu);                         // tile_dim1=rows | tile_dim2=0
    g1[5] = (int)(d0_stride & 0xffffffffu);                // tensor_dim0_stride lo
    g1[6] = (int)((d0_stride >> 32) & 0xffffu);            // stride hi | dim1_stride lo=0
    g1[7] = 0;
    v4i z = {0, 0, 0, 0};
#if __clang_major__ >= 23
    v8i z8 = {0, 0, 0, 0, 0, 0, 0, 0};
    __builtin_amdgcn_tensor_load_to_lds(g0, g1, z, z, z8, 0);
#else
    __builtin_amdgcn_tensor_load_to_lds(g0, g1, z, z, 0);
#endif
}
#endif  // HAVE_TDM

// ---------------------------------------------------------------- reductions
__global__ __launch_bounds__(256) void k_absum(const float* __restrict__ w,
                                               long long n, float* out) {
    __shared__ float red[256];
    long long stride = (long long)gridDim.x * blockDim.x;
    long long i = (long long)blockIdx.x * blockDim.x + threadIdx.x;
    float s = 0.f;
    for (; i < n; i += stride) s += fabsf(w[i]);
    red[threadIdx.x] = s;
    __syncthreads();
#pragma unroll
    for (int o = 128; o > 0; o >>= 1) {
        if ((int)threadIdx.x < o) red[threadIdx.x] += red[threadIdx.x + o];
        __syncthreads();
    }
    if (threadIdx.x == 0) atomicAdd(out, red[0]);
}

__global__ __launch_bounds__(256) void k_tern(const float* __restrict__ w,
                                              long long n,
                                              const float* __restrict__ sum,
                                              float invn,
                                              _Float16* __restrict__ out) {
    float am  = fmaxf(*sum * invn, 1e-5f);
    float thr = 0.7f * am;
    long long i = (long long)blockIdx.x * blockDim.x + threadIdx.x;
    if (i >= n) return;
    float v = w[i];
    float q = (v > thr) ? am : ((v < -thr) ? -am : 0.f);
    out[i] = (_Float16)q;
}

__global__ __launch_bounds__(256) void k_cvt(const float* __restrict__ x,
                                             long long n,
                                             _Float16* __restrict__ out) {
    long long i = (long long)blockIdx.x * blockDim.x + threadIdx.x;
    if (i < n) out[i] = (_Float16)x[i];
}

// ---------------------------------------------------------------- WMMA GEMM
// C[M,N](f32) = A[M,K](f16) @ W[N,K]^T(f16).  Block = 256 thr = 8 waves;
// block tile 128(M) x 64(N); wave tile 32x32 (4 WMMA accumulator chains).
// TDM path: wave 0 DMA-stages the 128x32 A tile and 64x32 W tile into LDS
// (double buffered, TENSORcnt-tracked); all waves consume via ds_load_b128.
__global__ __launch_bounds__(256) void k_gemm(const _Float16* __restrict__ A,
                                              const _Float16* __restrict__ W,
                                              float* __restrict__ C,
                                              int M, int N, int K) {
    int lane = threadIdx.x & 31, wave = threadIdx.x >> 5;
    int wm = wave & 3, wn = wave >> 2;
    int rb = blockIdx.y * 128 + wm * 32;
    int cb = blockIdx.x * 64 + wn * 32;
    int lm = lane & 15, hs = lane >> 4;

    v8f acc[2][2];
#pragma unroll
    for (int i = 0; i < 2; ++i)
#pragma unroll
        for (int j = 0; j < 2; ++j)
#pragma unroll
            for (int v = 0; v < 8; ++v) acc[i][j][v] = 0.f;

#if HAVE_TDM
    // 80B row stride (32 halves + TDM pad of 4 dwords), 16B aligned rows.
    __shared__ __align__(16) _Float16 lA[2][128 * 40];
    __shared__ __align__(16) _Float16 lB[2][64 * 40];
    const _Float16* Ab = A + (size_t)(blockIdx.y * 128) * K;
    const _Float16* Wb = W + (size_t)(blockIdx.x * 64) * K;

    if (wave == 0) {
        tdm_load_tile((unsigned)(uintptr_t)&lA[0][0], Ab, 128u, (unsigned long long)K);
        tdm_load_tile((unsigned)(uintptr_t)&lB[0][0], Wb, 64u, (unsigned long long)K);
    }
    int KT = K >> 5;
    int cur = 0;
    for (int kt = 0; kt < KT; ++kt) {
        if (wave == 0) __builtin_amdgcn_s_wait_tensorcnt(0);
        __syncthreads();  // staged tiles visible to all waves
        int nxt = cur ^ 1;
        if (wave == 0 && kt + 1 < KT) {
            int k0 = (kt + 1) << 5;
            tdm_load_tile((unsigned)(uintptr_t)&lA[nxt][0], Ab + k0, 128u,
                          (unsigned long long)K);
            tdm_load_tile((unsigned)(uintptr_t)&lB[nxt][0], Wb + k0, 64u,
                          (unsigned long long)K);
        }
        v16h a[2], b[2];
#pragma unroll
        for (int i = 0; i < 2; ++i) {
            const _Float16* p = &lA[cur][(wm * 32 + i * 16 + lm) * 40 + hs * 8];
            a[i] = mk16(p, p + 16);
        }
#pragma unroll
        for (int j = 0; j < 2; ++j) {
            const _Float16* p = &lB[cur][(wn * 32 + j * 16 + lm) * 40 + hs * 8];
            b[j] = mk16(p, p + 16);
        }
#pragma unroll
        for (int i = 0; i < 2; ++i)
#pragma unroll
            for (int j = 0; j < 2; ++j)
                acc[i][j] = wmma_f16(a[i], b[j], acc[i][j]);
        __syncthreads();  // all reads done before buffer is re-targeted
        cur = nxt;
    }
#else
    for (int k0 = 0; k0 < K; k0 += 32) {
        v16h a[2], b[2];
#pragma unroll
        for (int i = 0; i < 2; ++i) {
            const _Float16* p = A + (size_t)(rb + i * 16 + lm) * K + k0 + hs * 8;
            a[i] = mk16(p, p + 16);
        }
#pragma unroll
        for (int j = 0; j < 2; ++j) {
            const _Float16* p = W + (size_t)(cb + j * 16 + lm) * K + k0 + hs * 8;
            b[j] = mk16(p, p + 16);
        }
        if (k0 + 64 < K) {
            __builtin_prefetch(W + (size_t)(cb + lm) * K + k0 + 64, 0, 1);
        }
#pragma unroll
        for (int i = 0; i < 2; ++i)
#pragma unroll
            for (int j = 0; j < 2; ++j)
                acc[i][j] = wmma_f16(a[i], b[j], acc[i][j]);
    }
#endif

    // C/D layout: VGPR v -> row v (lanes 0-15) or v+8 (lanes 16-31); lane%16 -> col.
#pragma unroll
    for (int i = 0; i < 2; ++i)
#pragma unroll
        for (int j = 0; j < 2; ++j) {
            float* Cp = C + (size_t)(rb + i * 16 + hs * 8) * N + cb + j * 16 + lm;
#pragma unroll
            for (int v = 0; v < 8; ++v) Cp[(size_t)v * N] = acc[i][j][v];
        }
}

// ------------------------------------------------------- RoPE + layout repack
__global__ __launch_bounds__(256) void k_rope(const float* __restrict__ qkv,
                                              const float* __restrict__ cosp,
                                              const float* __restrict__ sinp,
                                              _Float16* __restrict__ qh,
                                              _Float16* __restrict__ kh,
                                              _Float16* __restrict__ vt,
                                              int B, int T, int H, int D) {
    long long n = (long long)B * T * H * D;
    long long i = (long long)blockIdx.x * blockDim.x + threadIdx.x;
    if (i >= n) return;
    int d = (int)(i % D);
    int h = (int)((i / D) % H);
    int t = (int)((i / ((long long)D * H)) % T);
    int b = (int)(i / ((long long)D * H * T));
    int Cc = H * D;
    size_t row = (size_t)(b * T + t) * (3 * Cc);
    int col = h * D + d;
    int dp = (d < D / 2) ? d + D / 2 : d - D / 2;
    float sg = (d < D / 2) ? -1.f : 1.f;
    float cs = cosp[(size_t)t * D + d];
    float sn = sinp[(size_t)t * D + d];

    float qv = qkv[row + col];
    float kv = qkv[row + Cc + col];
    float vv = qkv[row + 2 * Cc + col];
    float qp = qkv[row + h * D + dp];
    float kp = qkv[row + Cc + h * D + dp];

    float qo = qv * cs + sg * qp * sn;
    float ko = kv * cs + sg * kp * sn;

    size_t ht = (size_t)(b * H + h) * T + t;
    qh[ht * D + d] = (_Float16)qo;
    kh[ht * D + d] = (_Float16)ko;
    vt[((size_t)(b * H + h) * D + d) * T + t] = (_Float16)vv;  // V transposed
}

// -------------------------------------------------------- flash attention
__global__ __launch_bounds__(256) void k_flash(const _Float16* __restrict__ qh,
                                               const _Float16* __restrict__ kh,
                                               const _Float16* __restrict__ vt,
                                               _Float16* __restrict__ yh,
                                               int T, int H, int D) {
    __shared__ _Float16 plds[8 * 16 * 48];  // per-wave 16x32 P tile, stride 48
    int lane = threadIdx.x & 31, wave = threadIdx.x >> 5;
    int lm = lane & 15, hs = lane >> 4;
    int bh = blockIdx.x;
    int b = bh / H, h = bh % H;
    int qbase = blockIdx.y * 128 + wave * 16;

    const _Float16* Q  = qh + (size_t)bh * T * D;
    const _Float16* Kp = kh + (size_t)bh * T * D;
    const _Float16* Vt = vt + (size_t)bh * D * T;
    _Float16* lp = plds + wave * 16 * 48;

    v16h aq[2];
#pragma unroll
    for (int i = 0; i < 2; ++i) {
        const _Float16* p = Q + (size_t)(qbase + lm) * D + i * 32 + hs * 8;
        aq[i] = mk16(p, p + 16);
    }

    float m[8], l[8];
    v8f oc[4];
#pragma unroll
    for (int j = 0; j < 8; ++j) { m[j] = -1e30f; l[j] = 0.f; }
#pragma unroll
    for (int c = 0; c < 4; ++c)
#pragma unroll
        for (int v = 0; v < 8; ++v) oc[c][v] = 0.f;

    const float scale = 0.125f;  // 1/sqrt(64)
    int nkb = (qbase + 16 + 31) >> 5;  // causal: keys [0, qbase+16)

    for (int kb = 0; kb < nkb; ++kb) {
        int n0 = kb * 32;
        v8f s[2];
#pragma unroll
        for (int t = 0; t < 2; ++t)
#pragma unroll
            for (int v = 0; v < 8; ++v) s[t][v] = 0.f;

#pragma unroll
        for (int t = 0; t < 2; ++t) {
            int kr = n0 + t * 16 + lm;  // B col = key index
#pragma unroll
            for (int i = 0; i < 2; ++i) {
                const _Float16* p = Kp + (size_t)kr * D + i * 32 + hs * 8;
                v16h bk = mk16(p, p + 16);
                s[t] = wmma_f16(aq[i], bk, s[t]);
            }
        }

        float alpha[8];
#pragma unroll
        for (int j = 0; j < 8; ++j) {
            int rowq = qbase + j + hs * 8;
            float s0 = s[0][j] * scale;
            float s1 = s[1][j] * scale;
            if (n0 + lm > rowq)      s0 = -1e30f;
            if (n0 + 16 + lm > rowq) s1 = -1e30f;
            float mx = fmaxf(s0, s1);
#pragma unroll
            for (int off = 1; off < 16; off <<= 1)
                mx = fmaxf(mx, __shfl_xor(mx, off, 16));
            float mn = fmaxf(m[j], mx);
            float a  = __expf(m[j] - mn);
            float p0 = __expf(s0 - mn);
            float p1 = __expf(s1 - mn);
            float rs = p0 + p1;
#pragma unroll
            for (int off = 1; off < 16; off <<= 1)
                rs += __shfl_xor(rs, off, 16);
            l[j] = l[j] * a + rs;
            m[j] = mn;
            alpha[j] = a;
            int r = j + hs * 8;
            lp[r * 48 + lm]      = (_Float16)p0;
            lp[r * 48 + 16 + lm] = (_Float16)p1;
        }
#pragma unroll
        for (int c = 0; c < 4; ++c)
#pragma unroll
            for (int j = 0; j < 8; ++j) oc[c][j] *= alpha[j];

        asm volatile("" ::: "memory");  // order LDS store -> load (same wave)
        const _Float16* pp = lp + lm * 48 + hs * 8;
        v16h ap = mk16(pp, pp + 16);

#pragma unroll
        for (int c = 0; c < 4; ++c) {
            const _Float16* vp = Vt + (size_t)(c * 16 + lm) * T + n0 + hs * 8;
            v16h bv = mk16(vp, vp + 16);
            oc[c] = wmma_f16(ap, bv, oc[c]);
        }
    }

    int Cc = H * D;
#pragma unroll
    for (int c = 0; c < 4; ++c)
#pragma unroll
        for (int j = 0; j < 8; ++j) {
            int row = qbase + j + hs * 8;
            yh[(size_t)(b * T + row) * Cc + h * D + c * 16 + lm] =
                (_Float16)(oc[c][j] / l[j]);
        }
}

// ------------------------------------------------------------------- driver
extern "C" void kernel_launch(void* const* d_in, const int* in_sizes, int n_in,
                              void* d_out, int out_size, void* d_ws, size_t ws_size,
                              hipStream_t stream) {
    const float* x     = (const float*)d_in[0];
    const float* cosp  = (const float*)d_in[1];
    const float* sinp  = (const float*)d_in[2];
    const float* wqkv  = (const float*)d_in[3];
    const float* wproj = (const float*)d_in[4];
    float* out = (float*)d_out;

    const int B = 2, T = 2048, C = 1024, H = 16, D = 64;
    const int M = B * T;  // 4096

    char* ws = (char*)d_ws;
    size_t o = 0;
    float* sums = (float*)ws;                       o = 256;
    _Float16* xh  = (_Float16*)(ws + o);            o += (size_t)M * C * 2;
    _Float16* wqh = (_Float16*)(ws + o);            o += (size_t)3 * C * C * 2;
    _Float16* wph = (_Float16*)(ws + o);            o += (size_t)C * C * 2;
    float*    qkv = (float*)(ws + o);               o += (size_t)M * 3 * C * 4;
    _Float16* qh  = (_Float16*)(ws + o);            o += (size_t)M * C * 2;
    _Float16* kh  = (_Float16*)(ws + o);            o += (size_t)M * C * 2;
    _Float16* vt  = (_Float16*)(ws + o);            o += (size_t)M * C * 2;
    _Float16* yh  = (_Float16*)(ws + o);            o += (size_t)M * C * 2;

    hipMemsetAsync(sums, 0, 8, stream);

    long long nq = (long long)3 * C * C, np = (long long)C * C;
    k_absum<<<512, 256, 0, stream>>>(wqkv, nq, sums + 0);
    k_absum<<<512, 256, 0, stream>>>(wproj, np, sums + 1);
    k_tern<<<(int)((nq + 255) / 256), 256, 0, stream>>>(wqkv, nq, sums + 0,
                                                        1.f / (float)nq, wqh);
    k_tern<<<(int)((np + 255) / 256), 256, 0, stream>>>(wproj, np, sums + 1,
                                                        1.f / (float)np, wph);
    long long nx = (long long)M * C;
    k_cvt<<<(int)((nx + 255) / 256), 256, 0, stream>>>(x, nx, xh);

    // qkv = x @ wq^T   [4096 x 3072]
    k_gemm<<<dim3(3 * C / 64, M / 128), 256, 0, stream>>>(xh, wqh, qkv, M, 3 * C, C);

    long long nr = (long long)B * T * H * D;
    k_rope<<<(int)((nr + 255) / 256), 256, 0, stream>>>(qkv, cosp, sinp, qh, kh, vt,
                                                        B, T, H, D);

    // causal flash attention -> yh [B,T,C] f16
    k_flash<<<dim3(B * H, T / 128), 256, 0, stream>>>(qh, kh, vt, yh, T, H, D);

    // out = y @ wp^T   [4096 x 1024]
    k_gemm<<<dim3(C / 64, M / 128), 256, 0, stream>>>(yh, wph, out, M, C, C);
}
